// BiLSTMAttentionCRFModel_27917287424046
// MI455X (gfx1250) — matrix operand order repaired and checked by reading, not compiled, not run on hardware
//
#include <hip/hip_runtime.h>

// ---------- CDNA5 WMMA types ----------
typedef __attribute__((ext_vector_type(16))) __bf16 v16bf;
typedef __attribute__((ext_vector_type(8)))  float  v8f;
typedef __attribute__((ext_vector_type(4)))  unsigned int u32x4;

#define B_   128
#define T_   512
#define E_   300
#define EP_  320     // E padded to multiple of 32 (WMMA K)
#define HH_  128
#define G_   512     // 4*HH
#define H_   256
#define D_   32
#define K_   5
#define NT_  (B_*T_)

// Tensor Data Mover availability (probe-verified builtin on both toolchains,
// but with different arity: clang-22 (ROCm 7.2) = 5 args, clang-23 = 6 args).
#if defined(__has_builtin)
#  if __has_builtin(__builtin_amdgcn_tensor_load_to_lds) && \
      __has_builtin(__builtin_amdgcn_s_wait_tensorcnt)
#    define HAVE_TDM 1
#  endif
#endif
#ifndef HAVE_TDM
#  define HAVE_TDM 0
#endif

__device__ __forceinline__ unsigned short f2bf(float x) {
    unsigned int u = __float_as_uint(x);
    unsigned int r = (u + 0x7FFFu + ((u >> 16) & 1u)) >> 16; // RNE
    return (unsigned short)r;
}
__device__ __forceinline__ float bf2f(unsigned short h) {
    return __uint_as_float(((unsigned int)h) << 16);
}
__device__ __forceinline__ float sigm(float x) { return 1.0f / (1.0f + __expf(-x)); }

// Fragment assembly: 2x16B chunks reinterpreted as 16 bf16 lanes-worth
union Frag { u32x4 q[2]; v16bf bf; };

#if HAVE_TDM
typedef __attribute__((ext_vector_type(8))) int i32x8;
typedef __attribute__((ext_vector_type(4))) int i32x4;

// Issue a TDM 2D tile load: 16 rows x 512 f32, row stride in f32 elements.
// D# layout per CDNA5 ISA.
__device__ __forceinline__ void tdm_load_tile_f32(const float* gsrc, unsigned lds_off,
                                                  unsigned long long row_stride_elts) {
    unsigned long long ga = (unsigned long long)gsrc;
    u32x4 g0;
    g0.x = 1u;                                      // count=1, user D#
    g0.y = lds_off;                                 // lds_addr (bytes)
    g0.z = (unsigned)ga;                            // global_addr[31:0]
    g0.w = (unsigned)((ga >> 32) & 0x01FFFFFFu)     // global_addr[56:32]
         | (2u << 30);                              // type=2 ("image")
    i32x8 g1;
    g1[0] = (int)(2u << 16);                        // wg_mask=0, data_size=4B
    g1[1] = (int)((unsigned)(G_ & 0xFFFF) << 16);   // tensor_dim0[15:0] = 512
    g1[2] = 0;                                      // tensor_dim0 hi, tensor_dim1 lo
    g1[3] = (int)(16u | ((unsigned)G_ << 16));      // tensor_dim1 hi (big), tile_dim0=512
    g1[4] = 16;                                     // tile_dim1 = 16 rows, tile_dim2=0
    g1[5] = (int)(unsigned)(row_stride_elts & 0xFFFFFFFFu); // dim0_stride lo32
    g1[6] = (int)(unsigned)((row_stride_elts >> 32) & 0xFFFFu);
    g1[7] = 0;
    i32x4 z4 = {0, 0, 0, 0};
#if __clang_major__ >= 23
    i32x8 z8 = {0, 0, 0, 0, 0, 0, 0, 0};
    __builtin_amdgcn_tensor_load_to_lds(g0, g1, z4, z4, z8, 0);
#else
    __builtin_amdgcn_tensor_load_to_lds(g0, g1, z4, z4, 0);
#endif
}
#endif // HAVE_TDM

// ---------------------------------------------------------------------------
// K1: cast LSTM weights to bf16 (wih padded E->EP with zeros)
// ---------------------------------------------------------------------------
__global__ void cast_weights_kernel(const float* __restrict__ wih_f, const float* __restrict__ wih_b,
                                    const float* __restrict__ whh_f, const float* __restrict__ whh_b,
                                    unsigned short* wihf_bf, unsigned short* wihb_bf,
                                    unsigned short* whhf_bf, unsigned short* whhb_bf) {
    int idx = blockIdx.x * 256 + threadIdx.x;
    const int NWIH = G_ * EP_;   // 163840
    const int NWHH = G_ * HH_;   // 65536
    if (idx < NWIH) {
        int r = idx / EP_, c = idx - r * EP_;
        wihf_bf[idx] = (c < E_) ? f2bf(wih_f[r * E_ + c]) : (unsigned short)0;
    } else if (idx < 2 * NWIH) {
        int j = idx - NWIH; int r = j / EP_, c = j - r * EP_;
        wihb_bf[j] = (c < E_) ? f2bf(wih_b[r * E_ + c]) : (unsigned short)0;
    } else if (idx < 2 * NWIH + NWHH) {
        int j = idx - 2 * NWIH;
        whhf_bf[j] = f2bf(whh_f[j]);
    } else if (idx < 2 * NWIH + 2 * NWHH) {
        int j = idx - 2 * NWIH - NWHH;
        whhb_bf[j] = f2bf(whh_b[j]);
    }
}

// ---------------------------------------------------------------------------
// K2: embedding gather + cast to bf16 with zero padding to EP
// ---------------------------------------------------------------------------
__global__ void embed_gather_kernel(const int* __restrict__ tokens,
                                    const float* __restrict__ emb,
                                    unsigned short* __restrict__ ebf) {
    int idx = blockIdx.x * 256 + threadIdx.x;
    if (idx >= NT_ * EP_) return;
    int row = idx / EP_, col = idx - row * EP_;
    int tok = tokens[row];
    ebf[idx] = (col < E_) ? f2bf(emb[(size_t)tok * E_ + col]) : (unsigned short)0;
}

// ---------------------------------------------------------------------------
// K3: xp = embeds(bf16) @ wih^T(bf16) + bih + bhh   via v_wmma_f32_16x16x32_bf16
//   grid = (G_/64, NT_/128); 8 waves/block; each wave: 1 M-tile x 4 N-tiles,
//   A fragment held in VGPRs and reused across the 4 WMMAs per K-step.
// ---------------------------------------------------------------------------
__global__ void xp_gemm_kernel(const unsigned short* __restrict__ ebf,
                               const unsigned short* __restrict__ wbf,
                               const float* __restrict__ bih, const float* __restrict__ bhh,
                               float* __restrict__ xp) {
    int lane = threadIdx.x & 31;
    int wave = threadIdx.x >> 5;
    int mtile = blockIdx.y * 8 + wave;
    int nbase = blockIdx.x * 4;
    int lrow   = lane & 15;
    int hiHalf = (lane >> 4) & 1;
    int m = mtile * 16 + lrow;
    const unsigned short* arow = ebf + (size_t)m * EP_;

    const unsigned short* brow[4];
    float bias[4];
    int   ncol[4];
    v8f   acc[4];
#pragma unroll
    for (int w = 0; w < 4; ++w) {
        ncol[w] = (nbase + w) * 16 + lrow;
        brow[w] = wbf + (size_t)ncol[w] * EP_;
        bias[w] = bih[ncol[w]] + bhh[ncol[w]];
        acc[w]  = (v8f){0.f, 0.f, 0.f, 0.f, 0.f, 0.f, 0.f, 0.f};
    }

    for (int kb = 0; kb < EP_; kb += 32) {
        if (kb + 32 < EP_) __builtin_prefetch(arow + kb + 32, 0, 0);
        Frag a;
        int ka = kb + (hiHalf ? 8 : 0);
        a.q[0] = *(const u32x4*)(arow + ka);
        a.q[1] = *(const u32x4*)(arow + ka + 16);
        int kbb = kb + (hiHalf ? 8 : 0);
#pragma unroll
        for (int w = 0; w < 4; ++w) {
            Frag b;
            b.q[0] = *(const u32x4*)(brow[w] + kbb);
            b.q[1] = *(const u32x4*)(brow[w] + kbb + 16);
            acc[w] = __builtin_amdgcn_wmma_f32_16x16x32_bf16(false, a.bf, false, b.bf,
                                                             (short)0, acc[w], false, false);
        }
    }
    int mbase = mtile * 16 + (hiHalf ? 8 : 0);
#pragma unroll
    for (int w = 0; w < 4; ++w) {
        for (int r = 0; r < 8; ++r) {
            int mm = mbase + r;
            xp[(size_t)mm * G_ + ncol[w]] = acc[w][r] + bias[w];
        }
    }
}

// ---------------------------------------------------------------------------
// K4: one LSTM direction. Block = 16 batch rows, loops T internally.
//   Per step: g(16x512) = h(16x128,bf16,LDS) @ whh^T (WMMA, K=128) + xp[t]
//   xp[t] tiles are double-buffered into LDS by the Tensor Data Mover
//   (tensor_load_to_lds + s_wait_tensorcnt), g is formed in place over xp.
//   Dynamic LDS: h(4KB) + c(8KB) + 2x xp/g(32KB) = 76KB.
// ---------------------------------------------------------------------------
#define LSTM_SMEM_BYTES (16*HH_*2 + 16*HH_*4 + 2*16*G_*4)

__global__ void lstm_dir_kernel(const float* __restrict__ xp,
                                const unsigned short* __restrict__ whh_bf,
                                unsigned short* __restrict__ lstm_out,
                                int dir, int rev) {
    extern __shared__ __align__(16) char smem[];
    unsigned short* h_sm  = (unsigned short*)smem;                  // 16*HH bf16
    float*          c_sm  = (float*)(smem + 16 * HH_ * 2);          // 16*HH f32
    float*          xp_sm = (float*)(smem + 16 * HH_ * 2 + 16 * HH_ * 4); // [2][16*G_]

    int tid  = threadIdx.x;
    int lane = tid & 31, wave = tid >> 5;
    int bbase = blockIdx.x * 16;
    for (int i = tid; i < 16 * HH_; i += 256) { h_sm[i] = 0; c_sm[i] = 0.f; }
    __syncthreads();

    int lrow = lane & 15;
    int hiHalf = (lane >> 4) & 1;

#if HAVE_TDM
    // Prologue: stage xp tile for the first timestep into buffer 0.
    if (wave == 0) {
        int tt0 = rev ? (T_ - 1) : 0;
        tdm_load_tile_f32(xp + ((size_t)bbase * T_ + tt0) * G_,
                          (unsigned)(unsigned long long)(void*)xp_sm,
                          (unsigned long long)T_ * G_);
    }
#endif

    for (int t = 0; t < T_; ++t) {
        int tt  = rev ? (T_ - 1 - t) : t;
        int cur = t & 1;
        float* gbuf = xp_sm + cur * (16 * G_);

#if HAVE_TDM
        if (wave == 0) {
            if (t + 1 < T_) {
                int ttn = rev ? (T_ - 2 - t) : (t + 1);
                tdm_load_tile_f32(xp + ((size_t)bbase * T_ + ttn) * G_,
                                  (unsigned)(unsigned long long)(void*)(xp_sm + ((t + 1) & 1) * (16 * G_)),
                                  (unsigned long long)T_ * G_);
                __builtin_amdgcn_s_wait_tensorcnt(1); // oldest (cur) complete
            } else {
                __builtin_amdgcn_s_wait_tensorcnt(0);
            }
        }
        __syncthreads(); // cur buffer visible to all waves
#else
        for (int i = tid; i < 16 * G_; i += 256) {
            int mrow = i >> 9, col = i & (G_ - 1);
            gbuf[i] = xp[((size_t)(bbase + mrow) * T_ + tt) * G_ + col];
        }
        __syncthreads();
#endif
        // --- recurrent matmul: each wave computes 4 N-tiles of g (in place) ---
        for (int j = 0; j < 4; ++j) {
            int nt = wave * 4 + j;
            int n  = nt * 16 + lrow;
            const unsigned short* brow = whh_bf + (size_t)n * HH_;
            v8f acc = {0.f, 0.f, 0.f, 0.f, 0.f, 0.f, 0.f, 0.f};
            for (int kb = 0; kb < HH_; kb += 32) {
                Frag a, b;
                int ka = kb + (hiHalf ? 8 : 0);
                a.q[0] = *(const u32x4*)(h_sm + lrow * HH_ + ka);
                a.q[1] = *(const u32x4*)(h_sm + lrow * HH_ + ka + 16);
                int kbb = kb + (hiHalf ? 8 : 0);
                b.q[0] = *(const u32x4*)(brow + kbb);
                b.q[1] = *(const u32x4*)(brow + kbb + 16);
                acc = __builtin_amdgcn_wmma_f32_16x16x32_bf16(false, a.bf, false, b.bf,
                                                              (short)0, acc, false, false);
            }
            int mb = hiHalf ? 8 : 0;
            for (int r = 0; r < 8; ++r) {
                int mm = mb + r;
                int gi = mm * G_ + n;
                gbuf[gi] = acc[r] + gbuf[gi];   // g = h@whh^T + xp (in place)
            }
        }
        __syncthreads();
        // --- gates: 16*128 = 2048 elements over 256 threads ---
        for (int it = 0; it < 8; ++it) {
            int idx = tid + it * 256;
            int m = idx >> 7, jc = idx & 127;
            float gi = gbuf[m * G_ + jc];
            float gf = gbuf[m * G_ + 128 + jc];
            float gg = gbuf[m * G_ + 256 + jc];
            float go = gbuf[m * G_ + 384 + jc];
            float c  = sigm(gf) * c_sm[idx] + sigm(gi) * tanhf(gg);
            c_sm[idx] = c;
            float h  = sigm(go) * tanhf(c);
            unsigned short hb = f2bf(h);
            h_sm[idx] = hb;
            lstm_out[((size_t)(bbase + m) * T_ + tt) * H_ + dir * HH_ + jc] = hb;
        }
        __syncthreads();
    }
}

// ---------------------------------------------------------------------------
// K5: attention raw scores s[b,t] = lstm_out[b,t,:] . wa + ba   (wave per (b,t))
// ---------------------------------------------------------------------------
__global__ void attn_score_kernel(const unsigned short* __restrict__ lstm_out,
                                  const float* __restrict__ wa,
                                  const float* __restrict__ ba,
                                  float* __restrict__ s) {
    int lane = threadIdx.x & 31, wave = threadIdx.x >> 5;
    int idx = blockIdx.x * 8 + wave;
    const unsigned short* row = lstm_out + (size_t)idx * H_;
    union { u32x4 q; unsigned short h[8]; } uq;
    uq.q = *(const u32x4*)(row + lane * 8);
    float acc = 0.f;
    for (int i = 0; i < 8; ++i) acc += bf2f(uq.h[i]) * wa[lane * 8 + i];
    for (int off = 16; off; off >>= 1) acc += __shfl_xor(acc, off, 32);
    if (lane == 0) s[idx] = acc + ba[0];
}

// ---------------------------------------------------------------------------
// K6: softmax over time per batch row (block per b, 256 threads, T=512)
// ---------------------------------------------------------------------------
__global__ void softmax_time_kernel(const float* __restrict__ s, float* __restrict__ attnw) {
    __shared__ float red[256];
    int b = blockIdx.x, tid = threadIdx.x;
    float s0 = s[b * T_ + tid], s1 = s[b * T_ + tid + 256];
    red[tid] = fmaxf(s0, s1);
    __syncthreads();
    for (int off = 128; off; off >>= 1) { if (tid < off) red[tid] = fmaxf(red[tid], red[tid + off]); __syncthreads(); }
    float mx = red[0];
    __syncthreads();
    float e0 = __expf(s0 - mx), e1 = __expf(s1 - mx);
    red[tid] = e0 + e1;
    __syncthreads();
    for (int off = 128; off; off >>= 1) { if (tid < off) red[tid] += red[tid + off]; __syncthreads(); }
    float inv = 1.0f / red[0];
    attnw[b * T_ + tid] = e0 * inv;
    attnw[b * T_ + tid + 256] = e1 * inv;
}

// ---------------------------------------------------------------------------
// K7: emissions[b,t,:] = (attn*(lstm_out@w1^T) + b1) @ w2^T + b2
//   wave per (b,t); lane = hidden dim d (D=32 == wave32 lanes)
// ---------------------------------------------------------------------------
__global__ void emissions_kernel(const unsigned short* __restrict__ lstm_out,
                                 const float* __restrict__ attnw,
                                 const float* __restrict__ w1, const float* __restrict__ b1,
                                 const float* __restrict__ w2, const float* __restrict__ b2,
                                 float* __restrict__ em) {
    int lane = threadIdx.x & 31, wave = threadIdx.x >> 5;
    int idx = blockIdx.x * 8 + wave;
    const unsigned short* row = lstm_out + (size_t)idx * H_;
    const float* wrow = w1 + (size_t)lane * H_;
    float dotv = 0.f;
    for (int q = 0; q < H_; q += 4) {
        dotv += bf2f(row[q])     * wrow[q]
              + bf2f(row[q + 1]) * wrow[q + 1]
              + bf2f(row[q + 2]) * wrow[q + 2]
              + bf2f(row[q + 3]) * wrow[q + 3];
    }
    float hid = attnw[idx] * dotv + b1[lane];
    for (int k = 0; k < K_; ++k) {
        float p = hid * w2[k * D_ + lane];
        for (int off = 16; off; off >>= 1) p += __shfl_xor(p, off, 32);
        if (lane == 0) em[(size_t)idx * K_ + k] = p + b2[k];
    }
}

// ---------------------------------------------------------------------------
// K8: CRF NLL — one thread per batch row (128 threads, 1 block), scalar out
// ---------------------------------------------------------------------------
__global__ void crf_nll_kernel(const float* __restrict__ em, const int* __restrict__ tags,
                               const unsigned char* __restrict__ mask,
                               const float* __restrict__ start, const float* __restrict__ end,
                               const float* __restrict__ trans, float* __restrict__ out) {
    __shared__ float tr[25], st[5], en[5];
    __shared__ float red[128];
    int b = threadIdx.x;
    if (b < 25) tr[b] = trans[b];
    if (b < 5) { st[b] = start[b]; en[b] = end[b]; }
    __syncthreads();

    const float* emr = em + (size_t)b * T_ * K_;
    const int* tg = tags + b * T_;
    const unsigned char* mk = mask + b * T_;

    int prev = tg[0];
    float numer = st[prev] + emr[prev];
    float sc[5];
    for (int j = 0; j < 5; ++j) sc[j] = st[j] + emr[j];
    int cnt = mk[0] ? 1 : 0;

    for (int t = 1; t < T_; ++t) {
        int on = mk[t] ? 1 : 0;
        float mf = on ? 1.0f : 0.0f;
        int tgt = tg[t];
        numer += (tr[prev * 5 + tgt] + emr[t * K_ + tgt]) * mf;
        prev = tgt;
        cnt += on;
        float nxt[5];
        for (int j = 0; j < 5; ++j) {
            float mx = sc[0] + tr[0 * 5 + j];
            for (int i = 1; i < 5; ++i) mx = fmaxf(mx, sc[i] + tr[i * 5 + j]);
            float ssum = 0.f;
            for (int i = 0; i < 5; ++i) ssum += __expf(sc[i] + tr[i * 5 + j] - mx);
            nxt[j] = mx + __logf(ssum) + emr[t * K_ + j];
        }
        if (on) for (int j = 0; j < 5; ++j) sc[j] = nxt[j];
    }
    int lastIdx = cnt - 1; if (lastIdx < 0) lastIdx = 0;
    numer += en[tg[lastIdx]];
    float mx = sc[0] + en[0];
    for (int j = 1; j < 5; ++j) mx = fmaxf(mx, sc[j] + en[j]);
    float ssum = 0.f;
    for (int j = 0; j < 5; ++j) ssum += __expf(sc[j] + en[j] - mx);
    float denom = mx + __logf(ssum);

    red[b] = numer - denom;
    __syncthreads();
    for (int off = 64; off; off >>= 1) { if (b < off) red[b] += red[b + off]; __syncthreads(); }
    if (b == 0) out[0] = -red[0] / (float)B_;
}

// ---------------------------------------------------------------------------
extern "C" void kernel_launch(void* const* d_in, const int* in_sizes, int n_in,
                              void* d_out, int out_size, void* d_ws, size_t ws_size,
                              hipStream_t stream) {
    (void)in_sizes; (void)n_in; (void)out_size; (void)ws_size;
    const int*            tokens    = (const int*)d_in[0];
    const int*            tags      = (const int*)d_in[1];
    const unsigned char*  mask      = (const unsigned char*)d_in[2];
    const float*          emb       = (const float*)d_in[3];
    const float*          wih_f     = (const float*)d_in[4];
    const float*          whh_f     = (const float*)d_in[5];
    const float*          bih_f     = (const float*)d_in[6];
    const float*          bhh_f     = (const float*)d_in[7];
    const float*          wih_b     = (const float*)d_in[8];
    const float*          whh_b     = (const float*)d_in[9];
    const float*          bih_b     = (const float*)d_in[10];
    const float*          bhh_b     = (const float*)d_in[11];
    const float*          wa        = (const float*)d_in[12];
    const float*          ba        = (const float*)d_in[13];
    const float*          w1        = (const float*)d_in[14];
    const float*          b1        = (const float*)d_in[15];
    const float*          w2        = (const float*)d_in[16];
    const float*          b2        = (const float*)d_in[17];
    const float*          crf_start = (const float*)d_in[18];
    const float*          crf_end   = (const float*)d_in[19];
    const float*          crf_trans = (const float*)d_in[20];

    char* ws = (char*)d_ws;
    size_t off = 0;
    auto carve = [&](size_t bytes) -> char* {
        char* p = ws + off;
        off += (bytes + 255) & ~(size_t)255;
        return p;
    };
    unsigned short* ebf      = (unsigned short*)carve((size_t)NT_ * EP_ * 2);  // 40 MB
    unsigned short* wihf_bf  = (unsigned short*)carve((size_t)G_ * EP_ * 2);
    unsigned short* wihb_bf  = (unsigned short*)carve((size_t)G_ * EP_ * 2);
    unsigned short* whhf_bf  = (unsigned short*)carve((size_t)G_ * HH_ * 2);
    unsigned short* whhb_bf  = (unsigned short*)carve((size_t)G_ * HH_ * 2);
    float*          xpf      = (float*)carve((size_t)NT_ * G_ * 4);            // 128 MB
    float*          xpb      = (float*)carve((size_t)NT_ * G_ * 4);            // 128 MB
    unsigned short* lstm_out = (unsigned short*)carve((size_t)NT_ * H_ * 2);   // 32 MB
    float*          sraw     = (float*)carve((size_t)NT_ * 4);
    float*          attnw    = (float*)carve((size_t)NT_ * 4);
    float*          emis     = (float*)carve((size_t)NT_ * K_ * 4);

    // 1) weights -> bf16 (padded)
    {
        int total = 2 * G_ * EP_ + 2 * G_ * HH_;
        cast_weights_kernel<<<(total + 255) / 256, 256, 0, stream>>>(
            wih_f, wih_b, whh_f, whh_b, wihf_bf, wihb_bf, whhf_bf, whhb_bf);
    }
    // 2) embedding gather -> bf16 (padded)
    embed_gather_kernel<<<(NT_ * EP_ + 255) / 256, 256, 0, stream>>>(tokens, emb, ebf);
    // 3) input projections (big WMMA GEMMs, 4-way A reuse)
    {
        dim3 grid(G_ / 64, NT_ / 128);
        xp_gemm_kernel<<<grid, 256, 0, stream>>>(ebf, wihf_bf, bih_f, bhh_f, xpf);
        xp_gemm_kernel<<<grid, 256, 0, stream>>>(ebf, wihb_bf, bih_b, bhh_b, xpb);
    }
    // 4) recurrent LSTM, both directions (WMMA inner steps, TDM-staged xp)
    lstm_dir_kernel<<<B_ / 16, 256, LSTM_SMEM_BYTES, stream>>>(xpf, whhf_bf, lstm_out, 0, 0);
    lstm_dir_kernel<<<B_ / 16, 256, LSTM_SMEM_BYTES, stream>>>(xpb, whhb_bf, lstm_out, 1, 1);
    // 5) attention scores + softmax over time
    attn_score_kernel<<<NT_ / 8, 256, 0, stream>>>(lstm_out, wa, ba, sraw);
    softmax_time_kernel<<<B_, 256, 0, stream>>>(sraw, attnw);
    // 6) emissions
    emissions_kernel<<<NT_ / 8, 256, 0, stream>>>(lstm_out, attnw, w1, b1, w2, b2, emis);
    // 7) CRF NLL -> scalar
    crf_nll_kernel<<<1, B_, 0, stream>>>(emis, tags, mask, crf_start, crf_end, crf_trans,
                                         (float*)d_out);
}
